// MessagePassingConvolution_5265629905523
// MI455X (gfx1250) — matrix operand order repaired and checked by reading, not compiled, not run on hardware
//
#include <hip/hip_runtime.h>
#include <hip/hip_bf16.h>

// ---------------------------------------------------------------------------
// MACE-style message passing convolution for MI455X (gfx1250, wave32, WMMA).
//
// Edge-MLP (400k x [8->64->64->64->128]) runs on v_wmma_f32_16x16x32_bf16
// (bf16 A/B, f32 accum). Everything else fp32. segment_sum via
// global_atomic_add_f32 (output is 25.6MB -> fully L2 resident).
// silu uses v_exp_f32 + v_rcp_f32 (fast path) instead of IEEE division.
// ---------------------------------------------------------------------------

#define N_NODES   25000
#define N_EDGES   400000
#define CHANNELS  32
#define EFD       8       // edge feat dim
#define HID       64
#define OUTF      128     // CHANNELS * NUM_IRREPS

typedef __attribute__((ext_vector_type(16))) __bf16 v16bf;
typedef __attribute__((ext_vector_type(8)))  float  v8f;

// LDS plan (bf16 elements):
//   wt0 [64][32]  (K zero-padded 8->32)      2048
//   wt1 [64][64]                             4096
//   wt2 [64][64]                             4096
//   wt3 [128][64]                            8192
//   ea  [128][32] (edge feats, K padded)     4096
//   actA[128][64]                            8192
//   actB[128][64]                            8192
//   mix [128][128]                          16384
static constexpr int LDS_ELEMS = 2048 + 4096 + 4096 + 8192 + 4096 + 8192 + 8192 + 16384;
static constexpr int LDS_BYTES = LDS_ELEMS * 2;   // 110592 B

// Load a 16xbf16 fragment as two b128 LDS loads (two contiguous 8-elem runs).
__device__ __forceinline__ v16bf ldfrag(const __bf16* p0, const __bf16* p1) {
  union { v16bf v; uint4 q[2]; } u;
  u.q[0] = *(const uint4*)p0;
  u.q[1] = *(const uint4*)p1;
  return u.v;
}

__device__ __forceinline__ v8f wmma_bf16(v16bf a, v16bf b, v8f c) {
  // D = A(16x32) * B(32x16) + C, f32 accumulate
  return __builtin_amdgcn_wmma_f32_16x16x32_bf16(
      /*neg_a=*/false, a, /*neg_b=*/false, b,
      /*c_mod=*/(short)0, c, /*reuse_a=*/false, /*reuse_b=*/false);
}

// Fast silu: x * sigmoid(x) = x * rcp(1 + exp(-x)); v_exp_f32 + v_rcp_f32.
__device__ __forceinline__ float silu(float h) {
  return h * __builtin_amdgcn_rcpf(1.0f + __expf(-h));
}

// Hidden layer: src[128][64] bf16 -> dst[128][64] bf16, weights wt[N=64][K=64].
__device__ __forceinline__ void mlp_hidden_layer(
    const __bf16* src, const __bf16* wt, __bf16* dst,
    int erow, int wave16, int hi, int kbA, int kbB, int nlo)
{
  const __bf16* arow = src + erow * HID;
  // A fragments for K-chunks 0 and 1 (K=64 total)
  v16bf a0 = ldfrag(arow + kbA,      arow + 16 + kbA);
  v16bf a1 = ldfrag(arow + 32 + kbA, arow + 48 + kbA);
#pragma unroll
  for (int nt = 0; nt < 4; ++nt) {
    const __bf16* brow = wt + (nt * 16 + nlo) * HID;
    v16bf b0 = ldfrag(brow + kbB,      brow + kbB + 16);
    v16bf b1 = ldfrag(brow + 32 + kbB, brow + 48 + kbB);
    v8f acc = {};
    acc = wmma_bf16(a0, b0, acc);
    acc = wmma_bf16(a1, b1, acc);
#pragma unroll
    for (int r = 0; r < 8; ++r) {
      int m = r + 8 * hi;
      float h = acc[r] * 0.125f;              // fold W/sqrt(64)
      dst[(wave16 + m) * HID + nt * 16 + nlo] = (__bf16)silu(h);
    }
  }
}

__global__ void __launch_bounds__(256)
mace_msg_kernel(const float* __restrict__ node_feats,   // [N_NODES][32][4]
                const float* __restrict__ edge_attrs,   // [N_EDGES][4]
                const float* __restrict__ edge_feats,   // [N_EDGES][8]
                const int*   __restrict__ senders,
                const int*   __restrict__ receivers,
                const float* __restrict__ W0,           // [8][64]
                const float* __restrict__ W1,           // [64][64]
                const float* __restrict__ W2,           // [64][64]
                const float* __restrict__ W3,           // [64][128]
                float* __restrict__ out)                // [N_NODES][32][8]
{
  extern __shared__ char smem_raw[];
  __bf16* wt0  = (__bf16*)smem_raw;         // [64][32]
  __bf16* wt1  = wt0  + 64 * 32;            // [64][64]
  __bf16* wt2  = wt1  + 64 * 64;            // [64][64]
  __bf16* wt3  = wt2  + 64 * 64;            // [128][64]
  __bf16* ea   = wt3  + 128 * 64;           // [128][32]
  __bf16* actA = ea   + 128 * 32;           // [128][64]
  __bf16* actB = actA + 128 * 64;           // [128][64]
  __bf16* mixb = actB + 128 * 64;           // [128][128]

  const int tid   = threadIdx.x;
  const int eBase = blockIdx.x * 128;

  // ---- Phase A: stage weights (transposed, bf16) + edge feats into LDS ----
  for (int i = tid; i < 64 * 32; i += 256) {          // wt0: [n][k], pad K
    int n = i >> 5, k = i & 31;
    wt0[i] = (k < EFD) ? (__bf16)W0[k * 64 + n] : (__bf16)0.0f;
  }
  for (int i = tid; i < 64 * 64; i += 256) {          // wt1
    int n = i >> 6, k = i & 63;
    wt1[i] = (__bf16)W1[k * 64 + n];
  }
  for (int i = tid; i < 64 * 64; i += 256) {          // wt2
    int n = i >> 6, k = i & 63;
    wt2[i] = (__bf16)W2[k * 64 + n];
  }
  for (int i = tid; i < 128 * 64; i += 256) {         // wt3: [n][k], n<128
    int n = i >> 6, k = i & 63;
    wt3[i] = (__bf16)W3[k * OUTF + n];
  }
  for (int i = tid; i < 128 * 32; i += 256) {         // edge feats, pad K
    int el = i >> 5, k = i & 31;
    int e = eBase + el;
    float v = (k < EFD && e < N_EDGES) ? edge_feats[e * EFD + k] : 0.0f;
    ea[el * 32 + k] = (__bf16)v;
  }
  __syncthreads();

  // ---- Phase B: per-wave MLP over its 16 edges (all-lanes-active WMMA) ----
  const int lane   = tid & 31;
  const int wave   = tid >> 5;
  const int mrow   = lane & 15;       // A-matrix row within tile
  const int hi     = lane >> 4;       // lane half
  const int kbA    = hi * 8;          // A fragment K-base (ISA 16-bit A layout)
  const int kbB    = hi * 8;          // B fragment K-base (ISA 16-bit B layout)
  const int nlo    = lane & 15;       // B/D column within tile
  const int wave16 = wave * 16;
  const int erow   = wave16 + mrow;

  // Layer 0: ea[16][32] @ wt0 -> actA (scale 1/sqrt(8))
  {
    const __bf16* arow = ea + erow * 32;
    v16bf a = ldfrag(arow + kbA, arow + 16 + kbA);
#pragma unroll
    for (int nt = 0; nt < 4; ++nt) {
      const __bf16* brow = wt0 + (nt * 16 + nlo) * 32;
      v16bf b = ldfrag(brow + kbB, brow + kbB + 16);
      v8f acc = {};
      acc = wmma_bf16(a, b, acc);
#pragma unroll
      for (int r = 0; r < 8; ++r) {
        int m = r + 8 * hi;
        float h = acc[r] * 0.35355339059327373f;   // 1/sqrt(8)
        actA[(wave16 + m) * HID + nt * 16 + nlo] = (__bf16)silu(h);
      }
    }
  }

  mlp_hidden_layer(actA, wt1, actB, erow, wave16, hi, kbA, kbB, nlo);  // L1
  mlp_hidden_layer(actB, wt2, actA, erow, wave16, hi, kbA, kbB, nlo);  // L2

  // Layer 3: actA[16][64] @ wt3[128][64] -> mix bf16 (scale 1/8, no silu)
  {
    const __bf16* arow = actA + erow * HID;
    v16bf a0 = ldfrag(arow + kbA,      arow + 16 + kbA);
    v16bf a1 = ldfrag(arow + 32 + kbA, arow + 48 + kbA);
#pragma unroll
    for (int nt = 0; nt < 8; ++nt) {
      const __bf16* brow = wt3 + (nt * 16 + nlo) * HID;
      v16bf b0 = ldfrag(brow + kbB,      brow + kbB + 16);
      v16bf b1 = ldfrag(brow + 32 + kbB, brow + 48 + kbB);
      v8f acc = {};
      acc = wmma_bf16(a0, b0, acc);
      acc = wmma_bf16(a1, b1, acc);
#pragma unroll
      for (int r = 0; r < 8; ++r) {
        int m = r + 8 * hi;
        mixb[(wave16 + m) * OUTF + nt * 16 + nlo] = (__bf16)(acc[r] * 0.125f);
      }
    }
  }
  __syncthreads();

  // ---- Phase C: tensor-product messages + atomic scatter (segment_sum) ----
  // wave w, iter i handles edge (i*8 + w); lane = channel c (32 channels).
  const float4* nf4 = (const float4*)node_feats;
  const float4* at4 = (const float4*)edge_attrs;
  const float inv      = 0.25f;                  // 1/sqrt(16)
  const float rsqrt3   = 0.5773502691896258f;
#pragma unroll 1
  for (int iter = 0; iter < 16; ++iter) {
    int el = iter * 8 + wave;
    int e  = eBase + el;
    int c  = lane;
    if (e < N_EDGES) {
      int snd = senders[e];
      int rcv = receivers[e];
      float4 x  = nf4[(size_t)snd * 32 + c];     // s, v0, v1, v2
      float4 at = at4[e];                        // a_s, a_v0, a_v1, a_v2
      const __bf16* mp = mixb + el * OUTF + c * 4;
      float mix0 = (float)mp[0], mix1 = (float)mp[1];
      float mix2 = (float)mp[2], mix3 = (float)mp[3];

      float s  = x.x, v0 = x.y, v1 = x.z, v2 = x.w;
      float as = at.x, av0 = at.y, av1 = at.z, av2 = at.w;

      float o0 = s * as * mix0 * inv;
      float o1 = (v0 * av0 + v1 * av1 + v2 * av2) * rsqrt3 * mix1 * inv;
      float m2 = s  * mix2 * inv;   // times a_v[d]
      float m3 = as * mix3 * inv;   // times v[d]

      float* base = out + (size_t)rcv * 256 + c * 8;
      unsafeAtomicAdd(base + 0, o0);
      unsafeAtomicAdd(base + 1, o1);
      unsafeAtomicAdd(base + 2, m2 * av0);
      unsafeAtomicAdd(base + 3, m2 * av1);
      unsafeAtomicAdd(base + 4, m2 * av2);
      unsafeAtomicAdd(base + 5, v0 * m3);
      unsafeAtomicAdd(base + 6, v1 * m3);
      unsafeAtomicAdd(base + 7, v2 * m3);
    }
  }
}

__global__ void zero_out_kernel(float* __restrict__ out, int n) {
  int i = blockIdx.x * blockDim.x + threadIdx.x;
  if (i < n) out[i] = 0.0f;
}

extern "C" void kernel_launch(void* const* d_in, const int* in_sizes, int n_in,
                              void* d_out, int out_size, void* d_ws, size_t ws_size,
                              hipStream_t stream) {
  const float* node_feats = (const float*)d_in[0];
  const float* edge_attrs = (const float*)d_in[1];
  const float* edge_feats = (const float*)d_in[2];
  const int*   senders    = (const int*)  d_in[3];
  const int*   receivers  = (const int*)  d_in[4];
  const float* W0         = (const float*)d_in[5];
  const float* W1         = (const float*)d_in[6];
  const float* W2         = (const float*)d_in[7];
  const float* W3         = (const float*)d_in[8];
  float* out = (float*)d_out;

  // Output must be zeroed every launch (harness does not re-poison).
  zero_out_kernel<<<(out_size + 255) / 256, 256, 0, stream>>>(out, out_size);

  const int blocks = (N_EDGES + 127) / 128;   // 3125
  mace_msg_kernel<<<blocks, 256, LDS_BYTES, stream>>>(
      node_feats, edge_attrs, edge_feats, senders, receivers,
      W0, W1, W2, W3, out);
}
